// SOM2dLayer_40699110097526
// MI455X (gfx1250) — matrix operand order repaired and checked by reading, not compiled, not run on hardware
//
#include <hip/hip_runtime.h>
#include <hip/hip_bf16.h>
#include <math.h>

// ---------------------------------------------------------------------------
// SOM BMU search for MI455X (gfx1250, wave32, WMMA).
//   d2[b,n] = x2[b] - 2*(X W^T)[b,n] + w2[n]  ; argmin over n, sqrt(min).
// GEMM term done on bf16 matrix cores with 3-term split (hi*hi+hi*lo+lo*hi)
// for near-f32 accuracy; x2/w2 exact in f32. Compute-bound (AI ~2000 F/B);
// all weights (16MB bf16) live in the 192MB L2 across blocks.
// ---------------------------------------------------------------------------

typedef __attribute__((ext_vector_type(16))) __bf16 v16bf;
typedef __attribute__((ext_vector_type(8)))  float  v8f;

#define SOM_H     256
#define SOM_W     256
#define SOM_D     64
#define SOM_B     4096
#define SOM_N     (SOM_H * SOM_W)       // 65536 map cells
#define ROWS_PER_BLOCK 32               // two 16-row A tiles per block
#define WAVES_PER_BLOCK 8

// ---- workspace layout (bytes) --------------------------------------------
#define WS_WHI  0                                   // 65536*64 bf16 = 8 MB
#define WS_WLO  (WS_WHI + SOM_N * SOM_D * 2)        // 8 MB
#define WS_XHI  (WS_WLO + SOM_N * SOM_D * 2)        // 4096*64 bf16 = 512 KB
#define WS_XLO  (WS_XHI + SOM_B * SOM_D * 2)        // 512 KB
#define WS_W2   (WS_XLO + SOM_B * SOM_D * 2)        // 65536 f32 = 256 KB
#define WS_X2   (WS_W2  + SOM_N * 4)                // 4096 f32
// total ~17.3 MB of d_ws

// ---------------------------------------------------------------------------
// Prep: split f32 rows into bf16 hi/lo (RNE) and compute exact f32 row norms.
// One 64-thread block per row of length D=64.
// ---------------------------------------------------------------------------
__global__ void som_prep_kernel(const float* __restrict__ src,
                                __bf16* __restrict__ hi,
                                __bf16* __restrict__ lo,
                                float* __restrict__ sq) {
    const int row = blockIdx.x;
    const int t   = threadIdx.x;                 // 0..63
    __shared__ float s[64];
    const float  x = src[row * SOM_D + t];
    const __bf16 h = (__bf16)x;                  // RNE truncation to bf16
    const float  r = x - (float)h;               // residual, exactly representable path
    hi[row * SOM_D + t] = h;
    lo[row * SOM_D + t] = (__bf16)r;
    s[t] = x * x;
    __syncthreads();
    if (t == 0) {
        float acc = 0.f;
#pragma unroll
        for (int i = 0; i < 64; ++i) acc += s[i];
        sq[row] = acc;
    }
}

// ---------------------------------------------------------------------------
// Main kernel: each block owns 32 batch rows; 8 waves stripe the 4096 column
// tiles (16 cells each). Per tile: 4 B fragments (hi/lo x Klo/Khi) loaded
// straight into wave32 WMMA layout, 12 v_wmma_f32_16x16x32_bf16 (6 per row
// tile, split into two accumulator chains for ILP), fused running argmin in
// C-layout slots. Cross-lane reduce via shfl_xor inside 16-lane halves, then
// cross-wave reduce in LDS; finally sqrt + (y,x) decode.
// ---------------------------------------------------------------------------
__global__ void __launch_bounds__(256)
som_bmu_kernel(const __bf16* __restrict__ Whi, const __bf16* __restrict__ Wlo,
               const float* __restrict__ w2,
               const __bf16* __restrict__ Xhi, const __bf16* __restrict__ Xlo,
               const float* __restrict__ x2,
               int* __restrict__ bmu_out, float* __restrict__ qe_out) {
    const int row0 = blockIdx.x * ROWS_PER_BLOCK;
    const int tid  = threadIdx.x;
    const int lane = tid & 31;
    const int wave = tid >> 5;
    const int half = lane >> 4;          // 0: lanes 0-15, 1: lanes 16-31
    const int l16  = lane & 15;

    // ---- invariant A fragments (16-bit A 16x32 layout) --------------------
    // lane(0-15)=row M=l16, K 0..15; lanes 16-31: K 16..31 (matches B)
    const int ma   = row0 + l16;             // row-tile a
    const int mb   = row0 + 16 + l16;        // row-tile b
    const int koff = half * 16;
    v16bf ahi0, ahi1, alo0, alo1;            // tile a, K 0..31 / 32..63
    v16bf bhi0_, bhi1_, blo0_, blo1_;        // tile b  (named b* = row-tile b A frags)
#pragma unroll
    for (int e = 0; e < 8; ++e) {
        ahi0[e]   = Xhi[ma * SOM_D + koff + e];
        ahi0[e+8] = Xhi[ma * SOM_D + 8 + koff + e];
        ahi1[e]   = Xhi[ma * SOM_D + 32 + koff + e];
        ahi1[e+8] = Xhi[ma * SOM_D + 40 + koff + e];
        alo0[e]   = Xlo[ma * SOM_D + koff + e];
        alo0[e+8] = Xlo[ma * SOM_D + 8 + koff + e];
        alo1[e]   = Xlo[ma * SOM_D + 32 + koff + e];
        alo1[e+8] = Xlo[ma * SOM_D + 40 + koff + e];
        bhi0_[e]   = Xhi[mb * SOM_D + koff + e];
        bhi0_[e+8] = Xhi[mb * SOM_D + 8 + koff + e];
        bhi1_[e]   = Xhi[mb * SOM_D + 32 + koff + e];
        bhi1_[e+8] = Xhi[mb * SOM_D + 40 + koff + e];
        blo0_[e]   = Xlo[mb * SOM_D + koff + e];
        blo0_[e+8] = Xlo[mb * SOM_D + 8 + koff + e];
        blo1_[e]   = Xlo[mb * SOM_D + 32 + koff + e];
        blo1_[e+8] = Xlo[mb * SOM_D + 40 + koff + e];
    }

    float bva[8], bvb[8];
    int   bia[8], bib[8];
#pragma unroll
    for (int r = 0; r < 8; ++r) {
        bva[r] = 3.4e38f; bvb[r] = 3.4e38f; bia[r] = 0; bib[r] = 0;
    }

    // B (32x16), contiguous k order chosen here to match A: lanes 0-15 hold K=0..15 of col N=l16,
    // lanes 16-31 hold K=16..31 -> 16 contiguous bf16 = two b128 loads.
    const int kb = half * 16;
    for (int tile = wave; tile < SOM_N / 16; tile += WAVES_PER_BLOCK) {
        const int n = tile * 16 + l16;
        const __bf16* wh = Whi + (size_t)n * SOM_D;
        const __bf16* wl = Wlo + (size_t)n * SOM_D;
        v16bf Bh0 = *(const v16bf*)(wh + kb);         // K 0..31
        v16bf Bh1 = *(const v16bf*)(wh + 32 + kb);    // K 32..63
        v16bf Bl0 = *(const v16bf*)(wl + kb);
        v16bf Bl1 = *(const v16bf*)(wl + 32 + kb);
        const float w2n = w2[n];

        v8f ca1 = {}, ca2 = {}, cb1 = {}, cb2 = {};
        // row-tile a: hi*hi + hi*lo chain, lo*hi chain
        ca1 = __builtin_amdgcn_wmma_f32_16x16x32_bf16(false, ahi0, false, Bh0, (short)0, ca1, false, false);
        ca2 = __builtin_amdgcn_wmma_f32_16x16x32_bf16(false, alo0, false, Bh0, (short)0, ca2, false, false);
        cb1 = __builtin_amdgcn_wmma_f32_16x16x32_bf16(false, bhi0_, false, Bh0, (short)0, cb1, false, false);
        cb2 = __builtin_amdgcn_wmma_f32_16x16x32_bf16(false, blo0_, false, Bh0, (short)0, cb2, false, false);
        ca1 = __builtin_amdgcn_wmma_f32_16x16x32_bf16(false, ahi1, false, Bh1, (short)0, ca1, false, false);
        ca2 = __builtin_amdgcn_wmma_f32_16x16x32_bf16(false, alo1, false, Bh1, (short)0, ca2, false, false);
        cb1 = __builtin_amdgcn_wmma_f32_16x16x32_bf16(false, bhi1_, false, Bh1, (short)0, cb1, false, false);
        cb2 = __builtin_amdgcn_wmma_f32_16x16x32_bf16(false, blo1_, false, Bh1, (short)0, cb2, false, false);
        ca1 = __builtin_amdgcn_wmma_f32_16x16x32_bf16(false, ahi0, false, Bl0, (short)0, ca1, false, false);
        cb1 = __builtin_amdgcn_wmma_f32_16x16x32_bf16(false, bhi0_, false, Bl0, (short)0, cb1, false, false);
        ca1 = __builtin_amdgcn_wmma_f32_16x16x32_bf16(false, ahi1, false, Bl1, (short)0, ca1, false, false);
        cb1 = __builtin_amdgcn_wmma_f32_16x16x32_bf16(false, bhi1_, false, Bl1, (short)0, cb1, false, false);

        // running argmin in C layout: slot r = row (r + 8*half) of each tile
#pragma unroll
        for (int r = 0; r < 8; ++r) {
            const float va = fmaf(-2.f, ca1[r] + ca2[r], w2n);
            if (va < bva[r]) { bva[r] = va; bia[r] = n; }
            const float vb = fmaf(-2.f, cb1[r] + cb2[r], w2n);
            if (vb < bvb[r]) { bvb[r] = vb; bib[r] = n; }
        }
    }

    // ---- reduce across the 16 lanes of each half (xor masks stay in-half) --
#pragma unroll
    for (int r = 0; r < 8; ++r) {
        float va = bva[r]; int ia = bia[r];
        float vb = bvb[r]; int ib = bib[r];
#pragma unroll
        for (int msk = 1; msk <= 8; msk <<= 1) {
            float ova = __shfl_xor(va, msk, 32); int oia = __shfl_xor(ia, msk, 32);
            if (ova < va || (ova == va && oia < ia)) { va = ova; ia = oia; }
            float ovb = __shfl_xor(vb, msk, 32); int oib = __shfl_xor(ib, msk, 32);
            if (ovb < vb || (ovb == vb && oib < ib)) { vb = ovb; ib = oib; }
        }
        bva[r] = va; bia[r] = ia; bvb[r] = vb; bib[r] = ib;
    }

    // ---- cross-wave reduce via LDS ----------------------------------------
    __shared__ float s_val[WAVES_PER_BLOCK * ROWS_PER_BLOCK];
    __shared__ int   s_idx[WAVES_PER_BLOCK * ROWS_PER_BLOCK];
    if (l16 == 0) {
        const int base = wave * ROWS_PER_BLOCK + half * 8;
#pragma unroll
        for (int r = 0; r < 8; ++r) {
            s_val[base + r]      = bva[r];  s_idx[base + r]      = bia[r];
            s_val[base + 16 + r] = bvb[r];  s_idx[base + 16 + r] = bib[r];
        }
    }
    __syncthreads();

    if (tid < ROWS_PER_BLOCK) {
        float bv = 3.4e38f; int bi = 0;
#pragma unroll
        for (int w = 0; w < WAVES_PER_BLOCK; ++w) {
            const float ov = s_val[w * ROWS_PER_BLOCK + tid];
            const int   oi = s_idx[w * ROWS_PER_BLOCK + tid];
            if (ov < bv || (ov == bv && oi < bi)) { bv = ov; bi = oi; }
        }
        const int row = row0 + tid;
        float d2 = x2[row] + bv;                 // add back ||x||^2
        d2 = d2 > 0.f ? d2 : 0.f;                // numerical floor
        bmu_out[row * 2 + 0] = bi >> 8;          // y = n / 256
        bmu_out[row * 2 + 1] = bi & (SOM_W - 1); // x = n % 256
        qe_out[row] = sqrtf(d2);
    }
}

// ---------------------------------------------------------------------------
extern "C" void kernel_launch(void* const* d_in, const int* in_sizes, int n_in,
                              void* d_out, int out_size, void* d_ws, size_t ws_size,
                              hipStream_t stream) {
    (void)in_sizes; (void)n_in; (void)out_size; (void)ws_size;

    const float* inputs  = (const float*)d_in[0];   // [4096, 64]
    const float* weights = (const float*)d_in[1];   // [256, 256, 64]

    char* ws = (char*)d_ws;                         // needs ~17.3 MB
    __bf16* Whi = (__bf16*)(ws + WS_WHI);
    __bf16* Wlo = (__bf16*)(ws + WS_WLO);
    __bf16* Xhi = (__bf16*)(ws + WS_XHI);
    __bf16* Xlo = (__bf16*)(ws + WS_XLO);
    float*  w2  = (float*) (ws + WS_W2);
    float*  x2  = (float*) (ws + WS_X2);

    // Precision split + exact norms (bandwidth-trivial: ~3 passes over 17MB).
    som_prep_kernel<<<SOM_N, 64, 0, stream>>>(weights, Whi, Wlo, w2);
    som_prep_kernel<<<SOM_B, 64, 0, stream>>>(inputs,  Xhi, Xlo, x2);

    // Output: [4096,2] int32 BMU coords, then [4096] f32 quantization errors.
    int*   bmu = (int*)d_out;
    float* qe  = (float*)d_out + 2 * SOM_B;
    som_bmu_kernel<<<SOM_B / ROWS_PER_BLOCK, 256, 0, stream>>>(
        Whi, Wlo, w2, Xhi, Xlo, x2, bmu, qe);
}
